// FusedTransformerBlock_11579231830810
// MI455X (gfx1250) — matrix operand order repaired and checked by reading, not compiled, not run on hardware
//
#include <hip/hip_runtime.h>

// ---------------------------------------------------------------------------
// Types for CDNA5 WMMA (wave32)
// ---------------------------------------------------------------------------
typedef __attribute__((ext_vector_type(16))) __bf16        v16bf;
typedef __attribute__((ext_vector_type(8)))  float         v8f;
typedef __attribute__((ext_vector_type(16))) unsigned short v16u;
typedef __attribute__((ext_vector_type(8)))  unsigned short v8u;

__device__ __forceinline__ unsigned short f2bf(float f) {
  unsigned int u = __float_as_uint(f);
  u += 0x7FFFu + ((u >> 16) & 1u);           // round-to-nearest-even
  return (unsigned short)(u >> 16);
}

__device__ __forceinline__ v8f wmma_bf16(v16u a, v16u b, v8f c) {
  return __builtin_amdgcn_wmma_f32_16x16x32_bf16(
      false, __builtin_bit_cast(v16bf, a),
      false, __builtin_bit_cast(v16bf, b),
      (short)0, c, false, false);
}

// A-fragment (16x32 bf16, M x K). A pre-offset to tile row 0.
// lane L: row = L&15 ; K = s..s+15 (contiguous), s = (L>=16)?16:0
__device__ __forceinline__ v16u load_a_frag(const unsigned short* A, int lda,
                                            int k0, int lane) {
  const unsigned short* p =
      A + (size_t)(lane & 15) * lda + k0 + (((lane >> 4) & 1) ? 16 : 0);
  v8u lo = *(const v8u*)(p);
  v8u hi = *(const v8u*)(p + 8);
  v16u r;
#pragma unroll
  for (int i = 0; i < 8; ++i) { r[i] = lo[i]; r[8 + i] = hi[i]; }
  return r;
}

// B-fragment (32x16, K x N) where B = W^T, W row-major [N x K].
// lane L: col n = n0 + (L&15), K = k0 + ((L>=16)?16:0) .. +15 (contiguous)
__device__ __forceinline__ v16u load_bT_frag(const unsigned short* W, int ldw,
                                             int n0, int k0, int lane) {
  return *(const v16u*)(W + (size_t)(n0 + (lane & 15)) * ldw + k0 +
                        (((lane >> 4) & 1) ? 16 : 0));
}

// B-fragment from row-major B[k][n] (strided, used for V in P@V).
__device__ __forceinline__ v16u load_b_rm_frag(const unsigned short* Bm, int ldb,
                                               int k0, int n0, int lane) {
  const unsigned short* p =
      Bm + (size_t)(k0 + (((lane >> 4) & 1) ? 16 : 0)) * ldb + n0 + (lane & 15);
  v16u r;
#pragma unroll
  for (int i = 0; i < 16; ++i) r[i] = p[(size_t)i * ldb];
  return r;
}

// ---------------------------------------------------------------------------
// fp32 -> bf16 conversion (weights). n divisible by 256.
// ---------------------------------------------------------------------------
__global__ void tb_cvt_bf16(const float* __restrict__ in,
                            unsigned short* __restrict__ out) {
  size_t i = (size_t)blockIdx.x * 256 + threadIdx.x;
  out[i] = f2bf(in[i]);
}

// ---------------------------------------------------------------------------
// LayerNorm over D=1024, one row per block (256 threads, 4 elems each),
// bf16 output for WMMA consumption.
// ---------------------------------------------------------------------------
__global__ void tb_layernorm(const float* __restrict__ X,
                             const float* __restrict__ g,
                             const float* __restrict__ b,
                             unsigned short* __restrict__ Y) {
  const int D = 1024;
  int row = blockIdx.x;
  float4 v = ((const float4*)(X + (size_t)row * D))[threadIdx.x];
  float s  = v.x + v.y + v.z + v.w;
  float s2 = v.x * v.x + v.y * v.y + v.z * v.z + v.w * v.w;
#pragma unroll
  for (int o = 16; o; o >>= 1) {
    s  += __shfl_xor(s,  o, 32);
    s2 += __shfl_xor(s2, o, 32);
  }
  __shared__ float rs[8], rs2[8];
  int wave = threadIdx.x >> 5;
  if ((threadIdx.x & 31) == 0) { rs[wave] = s; rs2[wave] = s2; }
  __syncthreads();
  s = 0.f; s2 = 0.f;
#pragma unroll
  for (int i = 0; i < 8; ++i) { s += rs[i]; s2 += rs2[i]; }
  float mu   = s * (1.f / 1024.f);
  float var  = s2 * (1.f / 1024.f) - mu * mu;
  float rinv = rsqrtf(var + 1e-5f);
  int c = threadIdx.x * 4;
  unsigned short* y = Y + (size_t)row * D + c;
  float vv[4] = {v.x, v.y, v.z, v.w};
#pragma unroll
  for (int j = 0; j < 4; ++j)
    y[j] = f2bf((vv[j] - mu) * rinv * g[c + j] + b[c + j]);
}

// ---------------------------------------------------------------------------
// Register-blocked GEMM: each wave computes a 64x64 tile (4x4 WMMA tiles).
// Per k-step: 4 A-frags + 4 B-frags (8KB) -> 16 WMMAs (32 FLOP/byte).
// __launch_bounds__(256, 1): ~230 VGPRs needed; wave32 allows 1024/wave, so
// this still yields 4 waves/SIMD while eliminating accumulator spills.
// MODE 0: C(bf16) = A@W^T + bias
// MODE 1: C(f32)  = res + scale * (A@W^T + bias), scale = sigmoid(*gate) or 1
// ---------------------------------------------------------------------------
template <int MODE>
__global__ __launch_bounds__(256, 1)
void tb_gemm_blk(const unsigned short* __restrict__ A, int lda,
                 const unsigned short* __restrict__ W, int ldw,
                 const float* __restrict__ bias,
                 const float* __restrict__ res,
                 const float* __restrict__ gate,
                 void* __restrict__ Cv, int ldc,
                 int n64, int K) {
  int lane = threadIdx.x & 31;
  int t = blockIdx.x * 8 + (threadIdx.x >> 5);
  int m0 = (t / n64) * 64;
  int n0 = (t % n64) * 64;
  v8f zero = {};
  v8f acc[4][4];
#pragma unroll
  for (int i = 0; i < 4; ++i)
#pragma unroll
    for (int j = 0; j < 4; ++j) acc[i][j] = zero;

#pragma unroll 2
  for (int k0 = 0; k0 < K; k0 += 32) {
    v16u a[4], bb[4];
#pragma unroll
    for (int i = 0; i < 4; ++i)
      a[i] = load_a_frag(A + (size_t)(m0 + 16 * i) * lda, lda, k0, lane);
#pragma unroll
    for (int j = 0; j < 4; ++j)
      bb[j] = load_bT_frag(W, ldw, n0 + 16 * j, k0, lane);
#pragma unroll
    for (int i = 0; i < 4; ++i)
#pragma unroll
      for (int j = 0; j < 4; ++j)
        acc[i][j] = wmma_bf16(a[i], bb[j], acc[i][j]);
  }

  int cl = lane & 15;
  int rb = ((lane >> 4) & 1) ? 8 : 0;
  float sc = 1.f;
  if (MODE == 1 && gate) sc = 1.f / (1.f + __expf(-gate[0]));
#pragma unroll
  for (int i = 0; i < 4; ++i) {
#pragma unroll
    for (int j = 0; j < 4; ++j) {
      int col = n0 + 16 * j + cl;
      float bv = bias[col];
      int r0 = m0 + 16 * i + rb;
      if (MODE == 0) {
        unsigned short* C = (unsigned short*)Cv;
#pragma unroll
        for (int e = 0; e < 8; ++e)
          C[(size_t)(r0 + e) * ldc + col] = f2bf(acc[i][j][e] + bv);
      } else {
        float* C = (float*)Cv;
#pragma unroll
        for (int e = 0; e < 8; ++e) {
          size_t idx = (size_t)(r0 + e) * ldc + col;
          C[idx] = res[idx] + sc * (acc[i][j][e] + bv);
        }
      }
    }
  }
}

// ---------------------------------------------------------------------------
// Blocked fused GEGLU GEMM: 64 rows x 32 val-cols (+ matching gate cols),
// dual accumulators sharing A-fragments.
// g(bf16) = (A@W1[0:FF]^T+b1[0:FF]) * gelu(A@W1[FF:2FF]^T+b1[FF:2FF])
// ---------------------------------------------------------------------------
__global__ __launch_bounds__(256, 1)
void tb_gemm_geglu_blk(const unsigned short* __restrict__ A, int lda,
                       const unsigned short* __restrict__ W1, int ldw,
                       const float* __restrict__ b1, int ff,
                       unsigned short* __restrict__ C, int ldc,
                       int n32, int K) {
  int lane = threadIdx.x & 31;
  int t = blockIdx.x * 8 + (threadIdx.x >> 5);
  int m0 = (t / n32) * 64;
  int n0 = (t % n32) * 32;
  v8f zero = {};
  v8f av[4][2], ag[4][2];
#pragma unroll
  for (int i = 0; i < 4; ++i)
#pragma unroll
    for (int j = 0; j < 2; ++j) { av[i][j] = zero; ag[i][j] = zero; }

#pragma unroll 2
  for (int k0 = 0; k0 < K; k0 += 32) {
    v16u a[4], bv[2], bg[2];
#pragma unroll
    for (int i = 0; i < 4; ++i)
      a[i] = load_a_frag(A + (size_t)(m0 + 16 * i) * lda, lda, k0, lane);
#pragma unroll
    for (int j = 0; j < 2; ++j) {
      bv[j] = load_bT_frag(W1, ldw, n0 + 16 * j,      k0, lane);
      bg[j] = load_bT_frag(W1, ldw, ff + n0 + 16 * j, k0, lane);
    }
#pragma unroll
    for (int i = 0; i < 4; ++i)
#pragma unroll
      for (int j = 0; j < 2; ++j) {
        av[i][j] = wmma_bf16(a[i], bv[j], av[i][j]);
        ag[i][j] = wmma_bf16(a[i], bg[j], ag[i][j]);
      }
  }

  int cl = lane & 15;
  int rb = ((lane >> 4) & 1) ? 8 : 0;
#pragma unroll
  for (int i = 0; i < 4; ++i) {
#pragma unroll
    for (int j = 0; j < 2; ++j) {
      int col = n0 + 16 * j + cl;
      float bval = b1[col], bgat = b1[ff + col];
      int r0 = m0 + 16 * i + rb;
#pragma unroll
      for (int e = 0; e < 8; ++e) {
        float v = av[i][j][e] + bval;
        float g = ag[i][j][e] + bgat;
        float ge = 0.5f * g * (1.f + erff(g * 0.70710678118f)); // exact gelu
        C[(size_t)(r0 + e) * ldc + col] = f2bf(v * ge);
      }
    }
  }
}

// ---------------------------------------------------------------------------
// Flash attention, 32 q-rows (2 WMMA q-tiles) per wave so K-fragments and the
// strided V gathers are amortized over 2x the WMMAs. Streams keys in tiles of
// 32 with online softmax. HD = 64. bias (alibi) optional.
// ---------------------------------------------------------------------------
__global__ __launch_bounds__(256, 1)
void tb_flash(const unsigned short* __restrict__ Q, int ldq,
              const unsigned short* __restrict__ K, int ldk,
              const unsigned short* __restrict__ V, int ldv,
              const float* __restrict__ bias,
              unsigned short* __restrict__ O, int ldo,
              int Lq, int Lk, int kRows, int Hn) {
  __shared__ __align__(32) unsigned short plds[8][2][16 * 32];
  int lane = threadIdx.x & 31;
  int wave = threadIdx.x >> 5;
  int gwave = blockIdx.x * 8 + wave;
  int qt_total = Lq >> 5;                         // 32-row q tiles
  int b   = gwave / (Hn * qt_total);
  int rem = gwave % (Hn * qt_total);
  int h   = rem / qt_total;
  int qt  = rem % qt_total;

  const unsigned short* Qh = Q + (size_t)(b * Lq + qt * 32) * ldq + h * 64;
  const unsigned short* Kh = K + (size_t)(b * kRows) * ldk + h * 64;
  const unsigned short* Vh = V + (size_t)(b * kRows) * ldv + h * 64;
  const float* biasRow =
      bias ? bias + ((size_t)(b * Hn + h) * Lq + qt * 32) * Lk : (const float*)0;

  v16u qa[2][2];
#pragma unroll
  for (int qi = 0; qi < 2; ++qi) {
    qa[qi][0] = load_a_frag(Qh + (size_t)(qi * 16) * ldq, ldq, 0, lane);
    qa[qi][1] = load_a_frag(Qh + (size_t)(qi * 16) * ldq, ldq, 32, lane);
  }

  v8f zero = {};
  v8f o[2][4];
  float mrow[2][8], lrow[2][8];
#pragma unroll
  for (int qi = 0; qi < 2; ++qi) {
#pragma unroll
    for (int t = 0; t < 4; ++t) o[qi][t] = zero;
#pragma unroll
    for (int i = 0; i < 8; ++i) { mrow[qi][i] = -3.0e38f; lrow[qi][i] = 0.f; }
  }

  int col = lane & 15;
  int rbase = ((lane >> 4) & 1) ? 8 : 0;

  for (int kb = 0; kb < Lk; kb += 32) {
    v16u kf0 = load_bT_frag(Kh, ldk, kb,      0,  lane);
    v16u kf1 = load_bT_frag(Kh, ldk, kb,      32, lane);
    v16u kf2 = load_bT_frag(Kh, ldk, kb + 16, 0,  lane);
    v16u kf3 = load_bT_frag(Kh, ldk, kb + 16, 32, lane);

    v16u pa[2];
#pragma unroll
    for (int qi = 0; qi < 2; ++qi) {
      v8f s0 = zero, s1 = zero;
      s0 = wmma_bf16(qa[qi][0], kf0, s0);
      s0 = wmma_bf16(qa[qi][1], kf1, s0);
      s1 = wmma_bf16(qa[qi][0], kf2, s1);
      s1 = wmma_bf16(qa[qi][1], kf3, s1);

#pragma unroll
      for (int i = 0; i < 8; ++i) {
        float x0 = s0[i] * 0.125f;                  // 1/sqrt(64)
        float x1 = s1[i] * 0.125f;
        if (biasRow) {
          size_t br = (size_t)(qi * 16 + rbase + i) * Lk + kb + col;
          x0 += biasRow[br];
          x1 += biasRow[br + 16];
        }
        s0[i] = x0; s1[i] = x1;
      }

      float alpha[8];
#pragma unroll
      for (int i = 0; i < 8; ++i) {
        float mx = fmaxf(s0[i], s1[i]);
#pragma unroll
        for (int off = 8; off; off >>= 1) mx = fmaxf(mx, __shfl_xor(mx, off, 32));
        float mnew = fmaxf(mrow[qi][i], mx);
        alpha[i] = __expf(mrow[qi][i] - mnew);
        float p0 = __expf(s0[i] - mnew);
        float p1 = __expf(s1[i] - mnew);
        s0[i] = p0; s1[i] = p1;
        float ps = p0 + p1;
#pragma unroll
        for (int off = 8; off; off >>= 1) ps += __shfl_xor(ps, off, 32);
        lrow[qi][i] = lrow[qi][i] * alpha[i] + ps;
        mrow[qi][i] = mnew;
      }
#pragma unroll
      for (int i = 0; i < 8; ++i) {
        o[qi][0][i] *= alpha[i]; o[qi][1][i] *= alpha[i];
        o[qi][2][i] *= alpha[i]; o[qi][3][i] *= alpha[i];
      }
      // C-fragment (lane=col) -> A-fragment (lane=row) transpose via LDS
      unsigned short* pbuf = plds[wave][qi];
#pragma unroll
      for (int i = 0; i < 8; ++i) {
        pbuf[(rbase + i) * 32 + col]      = f2bf(s0[i]);
        pbuf[(rbase + i) * 32 + 16 + col] = f2bf(s1[i]);
      }
      pa[qi] = load_a_frag(pbuf, 32, 0, lane); // same-wave DS ops are in-order
    }

    v16u vb0 = load_b_rm_frag(Vh, ldv, kb, 0,  lane);
    v16u vb1 = load_b_rm_frag(Vh, ldv, kb, 16, lane);
    v16u vb2 = load_b_rm_frag(Vh, ldv, kb, 32, lane);
    v16u vb3 = load_b_rm_frag(Vh, ldv, kb, 48, lane);
#pragma unroll
    for (int qi = 0; qi < 2; ++qi) {
      o[qi][0] = wmma_bf16(pa[qi], vb0, o[qi][0]);
      o[qi][1] = wmma_bf16(pa[qi], vb1, o[qi][1]);
      o[qi][2] = wmma_bf16(pa[qi], vb2, o[qi][2]);
      o[qi][3] = wmma_bf16(pa[qi], vb3, o[qi][3]);
    }
  }

#pragma unroll
  for (int qi = 0; qi < 2; ++qi) {
    unsigned short* Oh =
        O + (size_t)(b * Lq + qt * 32 + qi * 16) * ldo + h * 64;
#pragma unroll
    for (int i = 0; i < 8; ++i) {
      float inv = 1.f / lrow[qi][i];
      size_t rr = (size_t)(rbase + i) * ldo;
      Oh[rr + col]      = f2bf(o[qi][0][i] * inv);
      Oh[rr + 16 + col] = f2bf(o[qi][1][i] * inv);
      Oh[rr + 32 + col] = f2bf(o[qi][2][i] * inv);
      Oh[rr + 48 + col] = f2bf(o[qi][3][i] * inv);
    }
  }
}

// ---------------------------------------------------------------------------
// Host orchestration
// ---------------------------------------------------------------------------
extern "C" void kernel_launch(void* const* d_in, const int* in_sizes, int n_in,
                              void* d_out, int out_size, void* d_ws, size_t ws_size,
                              hipStream_t stream) {
  (void)in_sizes; (void)n_in; (void)out_size; (void)ws_size;
  constexpr int Bc = 4, Lc = 1024, Dc = 1024, Hc = 16, FF_ = 4096, NSc = 64;
  const int M = Bc * Lc;   // 4096 query rows

  const float* x        = (const float*)d_in[0];
  const float* alibi    = (const float*)d_in[1];
  const float* species  = (const float*)d_in[2];
  const float* n1g      = (const float*)d_in[3];
  const float* n1b      = (const float*)d_in[4];
  const float* sa_wqkv  = (const float*)d_in[5];
  const float* sa_bqkv  = (const float*)d_in[6];
  const float* sa_wo    = (const float*)d_in[7];
  const float* sa_bo    = (const float*)d_in[8];
  const float* nqg      = (const float*)d_in[9];
  const float* nqb      = (const float*)d_in[10];
  const float* nkvg     = (const float*)d_in[11];
  const float* nkvb     = (const float*)d_in[12];
  const float* ca_wqkv  = (const float*)d_in[13];
  const float* ca_bqkv  = (const float*)d_in[14];
  const float* ca_wo    = (const float*)d_in[15];
  const float* ca_bo    = (const float*)d_in[16];
  const float* gatep    = (const float*)d_in[17];
  const float* fg       = (const float*)d_in[18];
  const float* fb       = (const float*)d_in[19];
  const float* w1       = (const float*)d_in[20];
  const float* b1       = (const float*)d_in[21];
  const float* w2       = (const float*)d_in[22];
  const float* b2       = (const float*)d_in[23];
  float* out = (float*)d_out;

  // ---- workspace bump allocator ----
  char* wsb = (char*)d_ws;
  size_t off = 0;
  auto alloc = [&](size_t bytes) -> void* {
    void* p = wsb + off;
    off = (off + bytes + 255) & ~(size_t)255;
    return p;
  };
  unsigned short* wb_sa_qkv = (unsigned short*)alloc((size_t)3 * Dc * Dc * 2);
  unsigned short* wb_sa_o   = (unsigned short*)alloc((size_t)Dc * Dc * 2);
  unsigned short* wb_ca_qkv = (unsigned short*)alloc((size_t)3 * Dc * Dc * 2);
  unsigned short* wb_ca_o   = (unsigned short*)alloc((size_t)Dc * Dc * 2);
  unsigned short* wb_w1     = (unsigned short*)alloc((size_t)2 * FF_ * Dc * 2);
  unsigned short* wb_w2     = (unsigned short*)alloc((size_t)Dc * FF_ * 2);
  unsigned short* lnbuf     = (unsigned short*)alloc((size_t)M * Dc * 2);   // reused 3x
  unsigned short* lnkv      = (unsigned short*)alloc((size_t)Bc * NSc * Dc * 2);
  unsigned short* qkvbuf    = (unsigned short*)alloc((size_t)M * 3 * Dc * 2); // also qx/kvx
  unsigned short* attnbuf   = (unsigned short*)alloc((size_t)M * Dc * 2);   // reused 2x
  float*          x1        = (float*)alloc((size_t)M * Dc * 4);
  float*          x2        = (float*)alloc((size_t)M * Dc * 4);
  unsigned short* gbuf      = (unsigned short*)alloc((size_t)M * FF_ * 2);

  auto cvt = [&](const float* src, unsigned short* dst, size_t n) {
    tb_cvt_bf16<<<(unsigned)(n / 256), 256, 0, stream>>>(src, dst);
  };
  // 1) weights -> bf16
  cvt(sa_wqkv, wb_sa_qkv, (size_t)3 * Dc * Dc);
  cvt(sa_wo,   wb_sa_o,   (size_t)Dc * Dc);
  cvt(ca_wqkv, wb_ca_qkv, (size_t)3 * Dc * Dc);
  cvt(ca_wo,   wb_ca_o,   (size_t)Dc * Dc);
  cvt(w1,      wb_w1,     (size_t)2 * FF_ * Dc);
  cvt(w2,      wb_w2,     (size_t)Dc * FF_);

  // 2) h = LN(x)
  tb_layernorm<<<M, 256, 0, stream>>>(x, n1g, n1b, lnbuf);
  // 3) qkv = h @ sa_wqkv^T + sa_bqkv   [M x 3D] bf16 (64x64 per wave)
  tb_gemm_blk<0><<<(M / 64) * (3 * Dc / 64) / 8, 256, 0, stream>>>(
      lnbuf, Dc, wb_sa_qkv, Dc, sa_bqkv, (const float*)0, (const float*)0,
      qkvbuf, 3 * Dc, 3 * Dc / 64, Dc);
  // 4) flash self-attention with alibi -> attnbuf bf16 [M x D]
  tb_flash<<<Bc * Hc * (Lc / 32) / 8, 256, 0, stream>>>(
      qkvbuf, 3 * Dc, qkvbuf + Dc, 3 * Dc, qkvbuf + 2 * Dc, 3 * Dc,
      alibi, attnbuf, Dc, Lc, Lc, Lc, Hc);
  // 5) x1 = x + attn @ sa_wo^T + sa_bo
  tb_gemm_blk<1><<<(M / 64) * (Dc / 64) / 8, 256, 0, stream>>>(
      attnbuf, Dc, wb_sa_o, Dc, sa_bo, x, (const float*)0,
      x1, Dc, Dc / 64, Dc);
  // 6) lnq = LN(x1) ; lnkv = LN(species)
  tb_layernorm<<<M, 256, 0, stream>>>(x1, nqg, nqb, lnbuf);
  tb_layernorm<<<Bc * NSc, 256, 0, stream>>>(species, nkvg, nkvb, lnkv);
  // 7) qx = lnq @ ca_wqkv[0:D]^T ; kvx = lnkv @ ca_wqkv[D:3D]^T
  unsigned short* qx  = qkvbuf;                       // [M x D]
  unsigned short* kvx = qkvbuf + (size_t)M * Dc;      // [B*NS x 2D]
  tb_gemm_blk<0><<<(M / 64) * (Dc / 64) / 8, 256, 0, stream>>>(
      lnbuf, Dc, wb_ca_qkv, Dc, ca_bqkv, (const float*)0, (const float*)0,
      qx, Dc, Dc / 64, Dc);
  tb_gemm_blk<0><<<(Bc * NSc / 64) * (2 * Dc / 64) / 8, 256, 0, stream>>>(
      lnkv, Dc, wb_ca_qkv + (size_t)Dc * Dc, Dc, ca_bqkv + Dc,
      (const float*)0, (const float*)0, kvx, 2 * Dc, 2 * Dc / 64, Dc);
  // 8) flash cross-attention (Lk = NS = 64, no bias)
  tb_flash<<<Bc * Hc * (Lc / 32) / 8, 256, 0, stream>>>(
      qx, Dc, kvx, 2 * Dc, kvx + Dc, 2 * Dc,
      (const float*)0, attnbuf, Dc, Lc, NSc, NSc, Hc);
  // 9) x2 = x1 + sigmoid(gate) * (cattn @ ca_wo^T + ca_bo)
  tb_gemm_blk<1><<<(M / 64) * (Dc / 64) / 8, 256, 0, stream>>>(
      attnbuf, Dc, wb_ca_o, Dc, ca_bo, x1, gatep, x2, Dc, Dc / 64, Dc);
  // 10) ln3 = LN(x2)
  tb_layernorm<<<M, 256, 0, stream>>>(x2, fg, fb, lnbuf);
  // 11) g = (ln3@W1a^T+b1a) * gelu(ln3@W1b^T+b1b)   [M x FF] bf16
  tb_gemm_geglu_blk<<<(M / 64) * (FF_ / 32) / 8, 256, 0, stream>>>(
      lnbuf, Dc, wb_w1, Dc, b1, FF_, gbuf, FF_, FF_ / 32, Dc);
  // 12) out = x2 + g @ w2^T + b2
  tb_gemm_blk<1><<<(M / 64) * (Dc / 64) / 8, 256, 0, stream>>>(
      gbuf, FF_, wb_w2, FF_, b2, x2, (const float*)0, out, Dc, Dc / 64, FF_);
}
